// BastileGptOssExperts_74801150427178
// MI455X (gfx1250) — compile-verified, matrix operand order checked by reading, not run on hardware
//
#include <hip/hip_runtime.h>
#include <hip/hip_bf16.h>
#include <stdint.h>

// ---------------------------------------------------------------------------
// GPT-OSS MoE experts for MI455X (gfx1250): bf16 WMMA 16x16x32 path.
// v2: weights pre-transposed+converted once (no hot-loop transpose),
//     register-buffered LDS pipeline, routing weight folded into act so
//     GEMM2 uses a single accumulator set across all experts.
// ---------------------------------------------------------------------------

typedef __bf16 bf16_t;
typedef __attribute__((ext_vector_type(16))) __bf16 v16bf;
typedef __attribute__((ext_vector_type(8)))  float  v8f;

#define TOKENS 4096
#define HID    1024
#define EDIM   1024
#define D2     2048
#define NE     8

// round-to-nearest-even f32 -> bf16 without relying on __bf16 arithmetic
__device__ __forceinline__ bf16_t f2bf(float x) {
  union { float f; unsigned u; } a; a.f = x;
  unsigned r = a.u + 0x7FFFu + ((a.u >> 16) & 1u);
  union { unsigned short s; bf16_t b; } c;
  c.s = (unsigned short)(r >> 16);
  return c.b;
}

// Build a bf16 WMMA fragment from a 32-element K-vector in LDS.
// Per the CDNA5 ISA 16-bit A/B layout, lane-half h needs K runs
// [h*8, h*8+8) and [16+h*8, 16+h*8+8): two aligned 16-byte LDS reads.
__device__ __forceinline__ v16bf load_frag(const bf16_t* kvec, int h) {
  union { uint4 u[2]; v16bf v; } f;
  f.u[0] = *(const uint4*)(kvec + h * 8);
  f.u[1] = *(const uint4*)(kvec + 16 + h * 8);
  return f.v;
}

// LDS staging: A tile 64 rows x 32 K, B tile 256 cols x 32 K (both K-major,
// thanks to pre-transposed weights). Row stride 40 bf16 (80B) keeps 16B
// alignment for b128 access and staggers banks.
struct Stage {
  bf16_t As[64 * 40];
  bf16_t Bs[256 * 40];
};
union SmemG1 {
  Stage s;
  float ep[8 * 16 * 64];   // per-wave 16x64 f32 epilogue staging
};

// Shared 2x4-fragment WMMA tile update (wave computes 32x64 of C).
__device__ __forceinline__ void wmma_tile(const bf16_t* As, const bf16_t* Bs,
                                          int wm, int wn, int ln, int h,
                                          v8f (&acc)[2][4]) {
  v16bf af[2], bfr[4];
  #pragma unroll
  for (int i = 0; i < 2; ++i)
    af[i] = load_frag(As + (wm * 32 + i * 16 + ln) * 40, h);
  #pragma unroll
  for (int j = 0; j < 4; ++j)
    bfr[j] = load_frag(Bs + (wn * 64 + j * 16 + ln) * 40, h);
  #pragma unroll
  for (int i = 0; i < 2; ++i)
    #pragma unroll
    for (int j = 0; j < 4; ++j)
      acc[i][j] = __builtin_amdgcn_wmma_f32_16x16x32_bf16(
          false, af[i], false, bfr[j], (short)0, acc[i][j], false, false);
}

// ---------------------------------------------------------------------------
__global__ __launch_bounds__(256)
void cvt_f32_to_bf16(const float* __restrict__ in, bf16_t* __restrict__ outp,
                     int n4) {
  int i = blockIdx.x * blockDim.x + threadIdx.x;
  int stride = gridDim.x * blockDim.x;
  for (; i < n4; i += stride) {
    float4 v = ((const float4*)in)[i];
    union { bf16_t b[4]; ushort4 u; } o;
    o.b[0] = f2bf(v.x); o.b[1] = f2bf(v.y);
    o.b[2] = f2bf(v.z); o.b[3] = f2bf(v.w);
    ((ushort4*)outp)[i] = o.u;
  }
}

// One-shot: W[e][K][N] f32 -> WT[e][N][K] bf16 (LDS-tiled, both sides coalesced)
__global__ __launch_bounds__(256)
void cvt_transpose_bf16(const float* __restrict__ W, bf16_t* __restrict__ WT,
                        int K, int N) {
  __shared__ bf16_t t[64 * 66];
  const int e  = blockIdx.z;
  const int k0 = blockIdx.y * 64;
  const int n0 = blockIdx.x * 64;
  const float* Wsrc = W  + (size_t)e * K * N;
  bf16_t*      Wdst = WT + (size_t)e * K * N;
  const int tid = threadIdx.x;
  const int rc = tid & 63, rr = tid >> 6;
  #pragma unroll
  for (int it = 0; it < 16; ++it) {
    int k = rr + it * 4;
    t[k * 66 + rc] = f2bf(Wsrc[(size_t)(k0 + k) * N + n0 + rc]);
  }
  __syncthreads();
  const int lane = tid & 31, wv = tid >> 5;
  #pragma unroll
  for (int q = 0; q < 8; ++q) {
    int n = wv + q * 8;
    union { bf16_t b[2]; unsigned u; } o;
    o.b[0] = t[(lane * 2)     * 66 + n];
    o.b[1] = t[(lane * 2 + 1) * 66 + n];
    *(unsigned*)(Wdst + (size_t)(n0 + n) * K + k0 + lane * 2) = o.u;
  }
}

// ---------------------------------------------------------------------------
// GEMM1: gate_up = X @ W1[e]; fused bias + clipped-SiLU gating; routing
// weight folded into the bf16 act output.
__global__ __launch_bounds__(256)
void moe_gemm1(const bf16_t* __restrict__ Xbf,
               const bf16_t* __restrict__ W1T,   // [e][2048 n][1024 k]
               const float*  __restrict__ b1,
               const float*  __restrict__ routing,
               bf16_t*       __restrict__ actbf) {
  __shared__ SmemG1 sm;
  const int tid  = threadIdx.x;
  const int wave = tid >> 5, lane = tid & 31;
  const int h = lane >> 4, ln = lane & 15;
  const int wm = wave & 1, wn = wave >> 1;
  const int e  = blockIdx.z;
  const int m0 = blockIdx.y * 64;
  const int n0 = blockIdx.x * 256;                 // gate_up column base
  const int tr = tid >> 2, tc = tid & 3;           // A staging: 4 thr/row

  const bf16_t* aSrc = Xbf + (size_t)(m0 + tr) * HID + tc * 8;
  const bf16_t* bSrc = W1T + (size_t)e * HID * D2 + (size_t)(n0 + tid) * HID;

  v8f acc[2][4] = {};
  uint4 aR, bR0, bR1, bR2, bR3;
  auto ldA = [&](int k0) { aR = *(const uint4*)(aSrc + k0); };
  auto ldB = [&](int k0) {
    const uint4* p = (const uint4*)(bSrc + k0);
    bR0 = p[0]; bR1 = p[1]; bR2 = p[2]; bR3 = p[3];
  };
  auto stA = [&]() { *(uint4*)&sm.s.As[tr * 40 + tc * 8] = aR; };
  auto stB = [&]() {
    uint4* q = (uint4*)&sm.s.Bs[tid * 40];
    q[0] = bR0; q[1] = bR1; q[2] = bR2; q[3] = bR3;
  };

  ldA(0); ldB(0); stA(); stB();
  for (int k0 = 0; k0 < HID; k0 += 32) {
    __syncthreads();                       // LDS chunk visible to all
    bool more = (k0 + 32) < HID;
    if (more) { ldA(k0 + 32); ldB(k0 + 32); }   // overlap with WMMAs
    wmma_tile(sm.s.As, sm.s.Bs, wm, wn, ln, h, acc);
    __syncthreads();                       // everyone done reading
    if (more) { stA(); stB(); }
  }

  // Epilogue: pair even(gate)/odd(up) cols via LDS, activation, fold routing.
  const int nActBase = (n0 >> 1) + wn * 32;
  const int gcol = n0 + wn * 64 + 2 * lane;
  const float bg = b1[(size_t)e * D2 + gcol];
  const float bu = b1[(size_t)e * D2 + gcol + 1];
  #pragma unroll
  for (int i = 0; i < 2; ++i) {
    __syncthreads();
    #pragma unroll
    for (int j = 0; j < 4; ++j)
      #pragma unroll
      for (int r = 0; r < 8; ++r)
        sm.ep[wave * 1024 + (r + 8 * h) * 64 + j * 16 + ln] = acc[i][j][r];
    __syncthreads();
    const int mBase = m0 + wm * 32 + i * 16;
    for (int m = 0; m < 16; ++m) {
      float g = sm.ep[wave * 1024 + m * 64 + 2 * lane]     + bg;
      float u = sm.ep[wave * 1024 + m * 64 + 2 * lane + 1] + bu;
      g = fminf(g, 7.0f);
      u = fminf(fmaxf(u, -7.0f), 7.0f);
      float glu = g / (1.0f + __expf(-1.702f * g));
      float rw = routing[(size_t)(mBase + m) * NE + e];   // uniform -> s_load
      actbf[((size_t)e * TOKENS + mBase + m) * EDIM + nActBase + lane] =
          f2bf((u + 1.0f) * glu * rw);
    }
  }
}

// ---------------------------------------------------------------------------
// GEMM2: out = sum_e (r.act[e]) @ W2[e] + sum_e r[t,e]*b2[e]; one accumulator
// set across the whole expert loop (K is effectively 8*1024).
__global__ __launch_bounds__(256)
void moe_gemm2(const bf16_t* __restrict__ actbf,
               const bf16_t* __restrict__ W2T,   // [e][1024 n][1024 k]
               const float*  __restrict__ b2,
               const float*  __restrict__ routing,
               float*        __restrict__ out) {
  __shared__ Stage sm;
  const int tid  = threadIdx.x;
  const int wave = tid >> 5, lane = tid & 31;
  const int h = lane >> 4, ln = lane & 15;
  const int wm = wave & 1, wn = wave >> 1;
  const int m0 = blockIdx.y * 64;
  const int n0 = blockIdx.x * 256;
  const int tr = tid >> 2, tc = tid & 3;

  v8f acc[2][4] = {};

  for (int e = 0; e < NE; ++e) {
    const bf16_t* aSrc = actbf + (size_t)e * TOKENS * EDIM +
                         (size_t)(m0 + tr) * EDIM + tc * 8;
    const bf16_t* bSrc = W2T + (size_t)e * EDIM * HID +
                         (size_t)(n0 + tid) * EDIM;
    uint4 aR, bR0, bR1, bR2, bR3;
    auto ldA = [&](int k0) { aR = *(const uint4*)(aSrc + k0); };
    auto ldB = [&](int k0) {
      const uint4* p = (const uint4*)(bSrc + k0);
      bR0 = p[0]; bR1 = p[1]; bR2 = p[2]; bR3 = p[3];
    };
    auto stA = [&]() { *(uint4*)&sm.As[tr * 40 + tc * 8] = aR; };
    auto stB = [&]() {
      uint4* q = (uint4*)&sm.Bs[tid * 40];
      q[0] = bR0; q[1] = bR1; q[2] = bR2; q[3] = bR3;
    };

    ldA(0); ldB(0); stA(); stB();   // safe: previous iter ended on a barrier
    for (int k0 = 0; k0 < EDIM; k0 += 32) {
      __syncthreads();
      bool more = (k0 + 32) < EDIM;
      if (more) { ldA(k0 + 32); ldB(k0 + 32); }
      wmma_tile(sm.As, sm.Bs, wm, wn, ln, h, acc);
      __syncthreads();
      if (more) { stA(); stB(); }
    }
  }

  // bias term: rank-8 outer product sum_e r[t,e]*b2[e][n]
  float bcol[4][NE];
  #pragma unroll
  for (int j = 0; j < 4; ++j)
    #pragma unroll
    for (int e = 0; e < NE; ++e)
      bcol[j][e] = b2[(size_t)e * HID + n0 + wn * 64 + j * 16 + ln];
  #pragma unroll
  for (int i = 0; i < 2; ++i)
    #pragma unroll
    for (int r = 0; r < 8; ++r) {
      int m = m0 + wm * 32 + i * 16 + r + 8 * h;
      #pragma unroll
      for (int e = 0; e < NE; ++e) {
        float rw = routing[(size_t)m * NE + e];
        #pragma unroll
        for (int j = 0; j < 4; ++j)
          acc[i][j][r] += rw * bcol[j][e];
      }
    }

  #pragma unroll
  for (int i = 0; i < 2; ++i)
    #pragma unroll
    for (int j = 0; j < 4; ++j)
      #pragma unroll
      for (int r = 0; r < 8; ++r)
        out[(size_t)(m0 + wm * 32 + i * 16 + r + 8 * h) * HID +
            n0 + wn * 64 + j * 16 + ln] = acc[i][j][r];
}

// ---------------------------------------------------------------------------
extern "C" void kernel_launch(void* const* d_in, const int* in_sizes, int n_in,
                              void* d_out, int out_size, void* d_ws, size_t ws_size,
                              hipStream_t stream) {
  const float* hs = (const float*)d_in[0];   // hidden_states [4096,1024]
  const float* rw = (const float*)d_in[1];   // routing_weights [4096,8]
  const float* w1 = (const float*)d_in[2];   // gate_up_proj [8,1024,2048]
  const float* b1 = (const float*)d_in[3];   // gate_up_proj_bias [8,2048]
  const float* w2 = (const float*)d_in[4];   // down_proj [8,1024,1024]
  const float* b2 = (const float*)d_in[5];   // down_proj_bias [8,1024]
  float* out = (float*)d_out;

  char* ws = (char*)d_ws;
  bf16_t* Xbf   = (bf16_t*)(ws);                              //  8 MiB
  bf16_t* W1T   = (bf16_t*)(ws + (size_t)8  * 1024 * 1024);   // 32 MiB
  bf16_t* W2T   = (bf16_t*)(ws + (size_t)40 * 1024 * 1024);   // 16 MiB
  bf16_t* Actbf = (bf16_t*)(ws + (size_t)56 * 1024 * 1024);   // 64 MiB

  cvt_f32_to_bf16<<<2048, 256, 0, stream>>>(hs, Xbf, (TOKENS * HID) / 4);
  cvt_transpose_bf16<<<dim3(D2 / 64, HID / 64, NE), 256, 0, stream>>>(
      w1, W1T, HID, D2);
  cvt_transpose_bf16<<<dim3(HID / 64, EDIM / 64, NE), 256, 0, stream>>>(
      w2, W2T, EDIM, HID);

  moe_gemm1<<<dim3(D2 / 256, TOKENS / 64, NE), 256, 0, stream>>>(
      Xbf, W1T, b1, rw, Actbf);
  moe_gemm2<<<dim3(HID / 256, TOKENS / 64), 256, 0, stream>>>(
      Actbf, W2T, b2, rw, out);
}
